// LiquidCore_61538291417819
// MI455X (gfx1250) — compile-verified, matrix operand order checked by reading, not run on hardware
//
#include <hip/hip_runtime.h>
#include <hip/hip_bf16.h>
#include <cstdint>
#include <cstddef>

// ---------------------------------------------------------------------------
// Problem constants (match reference)
// ---------------------------------------------------------------------------
#define DD     2048
#define VV     32000
#define HH     2048
#define DFF_   8192
#define LL     2
#define NSTEPS 8
#define TOPK   128
#define BB     4
#define TT     512
#define BT     (BB * TT)          // 2048

// GEMM tiling: block 64(M) x 256(N), K-tile 64, 8 waves, wave tile 32x64 (2x4 WMMA)
#define BM 64
#define BN 256
#define BK 64
#define LDS_STRIDE 72             // 64 bf16 + 8 pad (16B) -> DMA-padded, 16B aligned rows

typedef float  v8f   __attribute__((ext_vector_type(8)));
typedef __bf16 v16bf __attribute__((ext_vector_type(16)));
typedef unsigned int u32x4 __attribute__((ext_vector_type(4)));
typedef int si4 __attribute__((ext_vector_type(4)));
typedef int si8 __attribute__((ext_vector_type(8)));

union FragU { v16bf v; u32x4 q[2]; };

// fp32 -> bf16 round-to-nearest-even
__device__ __forceinline__ unsigned short f2bf(float f) {
    union { float f; unsigned u; } v; v.f = f;
    unsigned r = v.u + 0x7FFFu + ((v.u >> 16) & 1u);
    return (unsigned short)(r >> 16);
}

// Low 32 bits of a flat shared pointer = LDS byte offset (ISA aperture rule)
__device__ __forceinline__ unsigned lds_off(const void* p) {
    return (unsigned)(uintptr_t)p;
}

// ---------------------------------------------------------------------------
// Generic bf16 GEMM:  C[M,N] = A[M,K] * W[N,K]^T (+ bias[N])
// A, W bf16 row-major, K contiguous. C fp32. Requires N%BN==0, K%BK==0.
// A tile -> LDS via global_load_async_to_lds_b128 (ASYNCcnt).
// W tile -> LDS via one Tensor Data Mover op per K-tile (TENSORcnt), with the
// DMA itself emitting the bank-padded LDS layout (pad 16B every 128B).
// ---------------------------------------------------------------------------
__global__ __launch_bounds__(256)
void k_gemm_bf16(const unsigned short* __restrict__ A,
                 const unsigned short* __restrict__ W,
                 float* __restrict__ C,
                 const float* __restrict__ bias,
                 int M, int N, int K)
{
    __shared__ __attribute__((aligned(16))) unsigned short As[2][BM * LDS_STRIDE];
    __shared__ __attribute__((aligned(16))) unsigned short Ws[2][BN * LDS_STRIDE];

    const int tid   = threadIdx.x;
    const int lane  = tid & 31;
    const int wave  = tid >> 5;       // 0..7
    const int waveM = wave >> 2;      // 0..1  (rows of 32)
    const int waveN = wave & 3;       // 0..3  (cols of 64)
    const int bm    = blockIdx.y * BM;
    const int bn    = blockIdx.x * BN;

    v8f acc[2][4];
    #pragma unroll
    for (int i = 0; i < 2; ++i)
        #pragma unroll
        for (int j = 0; j < 4; ++j)
            #pragma unroll
            for (int e = 0; e < 8; ++e)
                acc[i][j][e] = 0.0f;

    // ---- zero-fill M-tail rows of both A buffers once (DMA skips them) ----
    if (bm + BM > M) {
        u32x4 z = {0u, 0u, 0u, 0u};
        for (int c = tid; c < (BM * BK / 8); c += 256) {
            int r = c >> 3, kc = (c & 7) << 3;
            if (bm + r >= M) {
                *(u32x4*)(void*)&As[0][r * LDS_STRIDE + kc] = z;
                *(u32x4*)(void*)&As[1][r * LDS_STRIDE + kc] = z;
            }
        }
    }

    // ---- A tile: per-lane async global->LDS copies (16B each) ----
    auto stageA = [&](int buf, int kt) {
        for (int c = tid; c < (BM * BK / 8); c += 256) {
            int r = c >> 3, kc = (c & 7) << 3;
            int gr = bm + r;
            if (gr < M) {
                unsigned laddr = lds_off(&As[buf][r * LDS_STRIDE + kc]);
                unsigned long long ga =
                    (unsigned long long)(uintptr_t)(A + (size_t)gr * K + kt + kc);
                asm volatile("global_load_async_to_lds_b128 %0, %1, off"
                             :: "v"(laddr), "v"(ga) : "memory");
            }
        }
    };

    // ---- W tile: one TDM descriptor, issued by wave 0 ----
    auto stageW = [&](int buf, int kt) {
        if (wave == 0) {
            unsigned long long ga =
                (unsigned long long)(uintptr_t)(W + (size_t)bn * K + kt);
            unsigned laddr = lds_off(&Ws[buf][0]);
            unsigned td0 = (unsigned)K;       // tensor dim0 (elements)
            unsigned td1 = (unsigned)N;       // tensor dim1
            unsigned long long st0 = (unsigned long long)(unsigned)K; // dim0 stride
            si4 g0;
            g0[0] = 1;                                              // count=1
            g0[1] = (int)laddr;                                     // lds_addr
            g0[2] = (int)(unsigned)(ga & 0xFFFFFFFFull);            // global_addr lo
            g0[3] = (int)((unsigned)((ga >> 32) & 0x01FFFFFFull)    // global_addr hi
                          | 0x80000000u);                           // type=2 ("image")
            si8 g1;
            // data_size=2B, pad_enable, pad_interval: 32 dwords (=4), pad_amount: 4 dwords (=3)
            g1[0] = (int)((1u << 16) | (1u << 20) | (4u << 22) | (3u << 25));
            g1[1] = (int)((td0 & 0xFFFFu) << 16);                   // tensor_dim0 lo16
            g1[2] = (int)((td0 >> 16) | ((td1 & 0xFFFFu) << 16));   // td0 hi | td1 lo
            g1[3] = (int)((td1 >> 16) | ((unsigned)BK << 16));      // td1 hi | tile_dim0
            g1[4] = (int)BN;                                        // tile_dim1 (tile_dim2=0)
            g1[5] = (int)(unsigned)(st0 & 0xFFFFFFFFull);           // stride0 lo32
            g1[6] = (int)(unsigned)((st0 >> 32) & 0xFFFFull);       // stride0 hi16
            g1[7] = 0;
            asm volatile("tensor_load_to_lds %0, %1" :: "s"(g0), "s"(g1) : "memory");
        }
    };

    // ---- 2-stage pipeline: DMA next tile while WMMA consumes current ----
    stageA(0, 0);
    stageW(0, 0);

    const int lr  = lane & 15;        // row/col within 16
    const int kb0 = (lane >> 4) << 3; // 0 or 8 (K sub-chunk per ISA bf16 layout)

    int cur = 0;
    for (int kt = 0; kt < K; kt += BK) {
        asm volatile("s_wait_asynccnt 0x0" ::: "memory");
        __builtin_amdgcn_s_wait_tensorcnt(0);
        __syncthreads();

        if (kt + BK < K) {
            stageA(cur ^ 1, kt + BK);
            stageW(cur ^ 1, kt + BK);
            if (kt + 2 * BK < K)   // speculative L2 prefetch of tile after next
                __builtin_prefetch(W + (size_t)(bn + tid) * K + kt + 2 * BK, 0, 0);
        }

        #pragma unroll
        for (int ks = 0; ks < BK; ks += 32) {
            FragU a[2], b[4];
            #pragma unroll
            for (int i = 0; i < 2; ++i) {
                int r = waveM * 32 + i * 16 + lr;
                const unsigned short* p = &As[cur][r * LDS_STRIDE + ks + kb0];
                a[i].q[0] = *(const u32x4*)(const void*)(p);
                a[i].q[1] = *(const u32x4*)(const void*)(p + 16);
            }
            #pragma unroll
            for (int j = 0; j < 4; ++j) {
                int r = waveN * 64 + j * 16 + lr;
                const unsigned short* p = &Ws[cur][r * LDS_STRIDE + ks + kb0];
                b[j].q[0] = *(const u32x4*)(const void*)(p);
                b[j].q[1] = *(const u32x4*)(const void*)(p + 16);
            }
            #pragma unroll
            for (int i = 0; i < 2; ++i)
                #pragma unroll
                for (int j = 0; j < 4; ++j)
                    acc[i][j] = __builtin_amdgcn_wmma_f32_16x16x32_bf16(
                        false, a[i].v, false, b[j].v,
                        (short)0, acc[i][j], false, false);
        }
        cur ^= 1;
    }

    // ---- store C per ISA f32 C/D layout: VGPR e -> M = e + 8*(lane>=16) ----
    const int rowOff = (lane >> 4) * 8;
    #pragma unroll
    for (int i = 0; i < 2; ++i) {
        #pragma unroll
        for (int j = 0; j < 4; ++j) {
            int gm0 = bm + waveM * 32 + i * 16 + rowOff;
            int gn  = bn + waveN * 64 + j * 16 + lr;
            float bb = bias ? bias[gn] : 0.0f;
            #pragma unroll
            for (int e = 0; e < 8; ++e) {
                int gm = gm0 + e;
                if (gm < M)
                    C[(size_t)gm * N + gn] = acc[i][j][e] + bb;
            }
        }
    }
}

// ---------------------------------------------------------------------------
// Elementwise / reduction kernels
// ---------------------------------------------------------------------------
__global__ void k_f2bf(const float* __restrict__ s, unsigned short* __restrict__ d, size_t n) {
    size_t i = (size_t)blockIdx.x * blockDim.x + threadIdx.x;
    if (i < n) d[i] = f2bf(s[i]);
}

// s: [K,N] row-major fp32 -> d: [N,K] row-major bf16 (transpose)
__global__ void k_f2bf_t(const float* __restrict__ s, unsigned short* __restrict__ d,
                         int K, int N) {
    size_t i = (size_t)blockIdx.x * blockDim.x + threadIdx.x;
    size_t total = (size_t)K * N;
    if (i >= total) return;
    int k = (int)(i / N);
    int n = (int)(i % N);
    d[(size_t)n * K + k] = f2bf(s[i]);
}

__global__ void k_gather(const int* __restrict__ ids, const float* __restrict__ emb,
                         float* __restrict__ x, int rows, int D) {
    size_t i = (size_t)blockIdx.x * blockDim.x + threadIdx.x;
    size_t total = (size_t)rows * D;
    if (i >= total) return;
    int r = (int)(i / D);
    int c = (int)(i % D);
    x[i] = emb[(size_t)ids[r] * D + c];
}

// z[r, 0:D] = bf16(a[r*astr + c]);  z[r, D:2D] = bf16(b[r*bstr + c])
__global__ void k_concat(const float* __restrict__ a, long astr,
                         const float* __restrict__ b, long bstr,
                         unsigned short* __restrict__ z, int rows, int D) {
    size_t i = (size_t)blockIdx.x * blockDim.x + threadIdx.x;
    size_t total = (size_t)rows * 2 * D;
    if (i >= total) return;
    int r = (int)(i / (2 * D));
    int c = (int)(i % (2 * D));
    float v = (c < D) ? a[(size_t)r * astr + c] : b[(size_t)r * bstr + (c - D)];
    z[i] = f2bf(v);
}

__global__ void k_silu_bf(const float* __restrict__ in, unsigned short* __restrict__ out,
                          size_t n) {
    size_t i = (size_t)blockIdx.x * blockDim.x + threadIdx.x;
    if (i >= n) return;
    float x = in[i];
    out[i] = f2bf(x / (1.0f + __expf(-x)));
}

__global__ void k_silu_mul(float* __restrict__ g, const float* __restrict__ u, size_t n) {
    size_t i = (size_t)blockIdx.x * blockDim.x + threadIdx.x;
    if (i >= n) return;
    float x = g[i];
    g[i] = (x / (1.0f + __expf(-x))) * u[i];
}

// rate = sigmoid(-fo); h' = rate*go + (1-rate)*(hin+ao); optional strided copy
__global__ void k_cfc_combine(const float* __restrict__ fo, const float* __restrict__ go,
                              const float* __restrict__ ao, const float* __restrict__ hin,
                              float* __restrict__ hout, float* __restrict__ extra,
                              long estr, int D, size_t n) {
    size_t i = (size_t)blockIdx.x * blockDim.x + threadIdx.x;
    if (i >= n) return;
    float rate = 1.0f / (1.0f + __expf(fo[i]));
    float hn = rate * go[i] + (1.0f - rate) * (hin[i] + ao[i]);
    hout[i] = hn;
    if (extra) {
        int r = (int)(i / D);
        int c = (int)(i % D);
        extra[(size_t)r * estr + c] = hn;
    }
}

// RMS norm: one block (256 thr) per row; out = bf16(x * rsqrt(mean(x^2)+eps) * w)
__global__ void k_rmsnorm(const float* __restrict__ x, const float* __restrict__ w,
                          unsigned short* __restrict__ out, int D) {
    int row = blockIdx.x;
    const float* xr = x + (size_t)row * D;
    __shared__ float sred[8];
    __shared__ float s_rs;
    float s = 0.0f;
    for (int i = threadIdx.x; i < D; i += blockDim.x) {
        float v = xr[i];
        s += v * v;
    }
    for (int o = 16; o > 0; o >>= 1) s += __shfl_xor(s, o, 32);
    if ((threadIdx.x & 31) == 0) sred[threadIdx.x >> 5] = s;
    __syncthreads();
    if (threadIdx.x == 0) {
        float t = 0.0f;
        for (int i = 0; i < 8; ++i) t += sred[i];
        s_rs = rsqrtf(t / (float)D + 1e-6f);
    }
    __syncthreads();
    float rs = s_rs;
    for (int i = threadIdx.x; i < D; i += blockDim.x)
        out[(size_t)row * D + i] = f2bf(xr[i] * rs * w[i]);
}

// Per-row k-th largest |act| via value bisection; one block per row
__global__ void k_topk_thr(const float* __restrict__ act, float* __restrict__ thr,
                           int F, int k) {
    int row = blockIdx.x;
    const float* a = act + (size_t)row * F;
    __shared__ float sred[8];
    __shared__ float s_lo, s_hi;
    float m = 0.0f;
    for (int i = threadIdx.x; i < F; i += blockDim.x) m = fmaxf(m, fabsf(a[i]));
    for (int o = 16; o > 0; o >>= 1) m = fmaxf(m, __shfl_xor(m, o, 32));
    if ((threadIdx.x & 31) == 0) sred[threadIdx.x >> 5] = m;
    __syncthreads();
    if (threadIdx.x == 0) {
        float mm = sred[0];
        for (int i = 1; i < 8; ++i) mm = fmaxf(mm, sred[i]);
        s_lo = 0.0f;
        s_hi = mm;
    }
    __syncthreads();
    for (int it = 0; it < 28; ++it) {
        float mid = 0.5f * (s_lo + s_hi);
        int cnt = 0;
        for (int i = threadIdx.x; i < F; i += blockDim.x) cnt += (fabsf(a[i]) >= mid);
        for (int o = 16; o > 0; o >>= 1) cnt += __shfl_xor(cnt, o, 32);
        __syncthreads();
        if ((threadIdx.x & 31) == 0) sred[threadIdx.x >> 5] = (float)cnt;
        __syncthreads();
        if (threadIdx.x == 0) {
            int c = 0;
            for (int i = 0; i < 8; ++i) c += (int)sred[i];
            if (c >= k) s_lo = mid; else s_hi = mid;
        }
        __syncthreads();
    }
    if (threadIdx.x == 0) thr[row] = s_lo;
}

__global__ void k_mask_topk(const float* __restrict__ act, const float* __restrict__ thr,
                            unsigned short* __restrict__ out, int F, size_t n) {
    size_t i = (size_t)blockIdx.x * blockDim.x + threadIdx.x;
    if (i >= n) return;
    int r = (int)(i / F);
    float v = act[i];
    out[i] = (fabsf(v) >= thr[r]) ? f2bf(v) : (unsigned short)0;
}

// ---------------------------------------------------------------------------
// Host orchestration
// ---------------------------------------------------------------------------
static inline unsigned cdiv(size_t a, unsigned b) { return (unsigned)((a + b - 1) / b); }

extern "C" void kernel_launch(void* const* d_in, const int* in_sizes, int n_in,
                              void* d_out, int out_size, void* d_ws, size_t ws_size,
                              hipStream_t stream) {
    (void)in_sizes; (void)n_in; (void)out_size; (void)ws_size;

    const int*   input_ids  = (const int*)  d_in[0];
    const float* embed      = (const float*)d_in[1];
    const float* final_norm = (const float*)d_in[2];
    const float* query_norm = (const float*)d_in[3];
    const float* W1f32[3] = { (const float*)d_in[4],  (const float*)d_in[8],  (const float*)d_in[12] }; // f,g,a
    const float* b1f32[3] = { (const float*)d_in[5],  (const float*)d_in[9],  (const float*)d_in[13] };
    const float* W2f32[3] = { (const float*)d_in[6],  (const float*)d_in[10], (const float*)d_in[14] };
    const float* b2f32[3] = { (const float*)d_in[7],  (const float*)d_in[11], (const float*)d_in[15] };
    const float* sdm_gate  = (const float*)d_in[16];
    const float* sdm_up    = (const float*)d_in[17];
    const float* sdm_down  = (const float*)d_in[18];

    // ---- bump allocator over d_ws ----
    char* ws = (char*)d_ws;
    size_t off = 0;
    auto alloc = [&](size_t bytes) -> void* {
        off = (off + 255) & ~(size_t)255;
        void* p = ws + off;
        off += bytes;
        return p;
    };

    unsigned short* W1bf[3]; unsigned short* W2bf[3];
    for (int m = 0; m < 3; ++m) {
        W1bf[m] = (unsigned short*)alloc((size_t)2 * DD * HH * 2);
        W2bf[m] = (unsigned short*)alloc((size_t)HH * DD * 2);
    }
    unsigned short* gatebf = (unsigned short*)alloc((size_t)LL * DFF_ * DD * 2);
    unsigned short* upbf   = (unsigned short*)alloc((size_t)LL * DFF_ * DD * 2);
    unsigned short* downbf = (unsigned short*)alloc((size_t)LL * DD * DFF_ * 2);
    unsigned short* embbf  = (unsigned short*)alloc((size_t)VV * DD * 2);

    float* x      = (float*)alloc((size_t)BT * DD * 4);   // [B,T,D] embedded input
    float* state  = (float*)alloc((size_t)BT * DD * 4);   // [B,T,D] CfC state
    float* hbuf   = (float*)alloc((size_t)BB * DD * 4);   // scan hidden [B,D]

    unsigned short* zs    = (unsigned short*)alloc((size_t)BB * 2 * DD * 2);
    float*          hids  = (float*)alloc((size_t)BB * HH * 4);
    unsigned short* hbfs  = (unsigned short*)alloc((size_t)BB * HH * 2);
    float* fos = (float*)alloc((size_t)BB * DD * 4);
    float* gos = (float*)alloc((size_t)BB * DD * 4);
    float* aos = (float*)alloc((size_t)BB * DD * 4);

    unsigned short* qbf   = (unsigned short*)alloc((size_t)BT * DD * 2);
    float*          gbuf  = (float*)alloc((size_t)BT * DFF_ * 4);
    float*          ubuf  = (float*)alloc((size_t)BT * DFF_ * 4);
    float*          thr   = (float*)alloc((size_t)BT * 4);
    unsigned short* actbf = (unsigned short*)alloc((size_t)BT * DFF_ * 2);
    float*          retr  = (float*)alloc((size_t)BT * DD * 4);

    unsigned short* zb    = (unsigned short*)alloc((size_t)BT * 2 * DD * 2);
    float*          hidb  = (float*)alloc((size_t)BT * HH * 4);
    unsigned short* hbfb  = (unsigned short*)alloc((size_t)BT * HH * 2);
    float* fob = (float*)alloc((size_t)BT * DD * 4);
    float* gob = (float*)alloc((size_t)BT * DD * 4);
    float* aob = (float*)alloc((size_t)BT * DD * 4);
    unsigned short* finbf = (unsigned short*)alloc((size_t)BT * DD * 2);

    // ---- launch helpers ----
    auto cvt = [&](const float* s, unsigned short* d, size_t n) {
        k_f2bf<<<cdiv(n, 256), 256, 0, stream>>>(s, d, n);
    };
    auto cvtT = [&](const float* s, unsigned short* d, int K, int N) {
        size_t n = (size_t)K * N;
        k_f2bf_t<<<cdiv(n, 256), 256, 0, stream>>>(s, d, K, N);
    };
    auto gemm = [&](const unsigned short* A, const unsigned short* Wm, float* C,
                    const float* bias, int M, int N, int K) {
        dim3 grid((unsigned)(N / BN), (unsigned)((M + BM - 1) / BM));
        k_gemm_bf16<<<grid, 256, 0, stream>>>(A, Wm, C, bias, M, N, K);
    };

    // ---- weight conversion (W1/W2 are [K,N] -> transpose to [N,K]) ----
    for (int m = 0; m < 3; ++m) {
        cvtT(W1f32[m], W1bf[m], 2 * DD, HH);
        cvtT(W2f32[m], W2bf[m], HH, DD);
    }
    cvt(sdm_gate, gatebf, (size_t)LL * DFF_ * DD);
    cvt(sdm_up,   upbf,   (size_t)LL * DFF_ * DD);
    cvt(sdm_down, downbf, (size_t)LL * DD * DFF_);
    cvt(embed,    embbf,  (size_t)VV * DD);

    // ---- embed gather ----
    k_gather<<<cdiv((size_t)BT * DD, 256), 256, 0, stream>>>(input_ids, embed, x, BT, DD);

    // ---- sequential CfC scan over T ----
    hipMemsetAsync(hbuf, 0, (size_t)BB * DD * 4, stream);
    for (int t = 0; t < TT; ++t) {
        size_t nz = (size_t)BB * 2 * DD;
        k_concat<<<cdiv(nz, 256), 256, 0, stream>>>(hbuf, DD, x + (size_t)t * DD,
                                                    (long)TT * DD, zs, BB, DD);
        float* outs[3] = { fos, gos, aos };
        for (int m = 0; m < 3; ++m) {
            gemm(zs, W1bf[m], hids, b1f32[m], BB, HH, 2 * DD);
            k_silu_bf<<<cdiv((size_t)BB * HH, 256), 256, 0, stream>>>(hids, hbfs, (size_t)BB * HH);
            gemm(hbfs, W2bf[m], outs[m], b2f32[m], BB, DD, HH);
        }
        k_cfc_combine<<<cdiv((size_t)BB * DD, 256), 256, 0, stream>>>(
            fos, gos, aos, hbuf, hbuf, state + (size_t)t * DD, (long)TT * DD, DD,
            (size_t)BB * DD);
    }

    // ---- 8 SDM + CfC refinement steps ----
    for (int s = 0; s < NSTEPS; ++s) {
        int layer = (s * LL) / NSTEPS;
        const unsigned short* gw = gatebf + (size_t)layer * DFF_ * DD;
        const unsigned short* uw = upbf   + (size_t)layer * DFF_ * DD;
        const unsigned short* dw = downbf + (size_t)layer * DD * DFF_;

        k_rmsnorm<<<BT, 256, 0, stream>>>(state, query_norm, qbf, DD);
        gemm(qbf, gw, gbuf, nullptr, BT, DFF_, DD);
        gemm(qbf, uw, ubuf, nullptr, BT, DFF_, DD);
        k_silu_mul<<<cdiv((size_t)BT * DFF_, 256), 256, 0, stream>>>(gbuf, ubuf,
                                                                     (size_t)BT * DFF_);
        k_topk_thr<<<BT, 256, 0, stream>>>(gbuf, thr, DFF_, TOPK);
        k_mask_topk<<<cdiv((size_t)BT * DFF_, 256), 256, 0, stream>>>(
            gbuf, thr, actbf, DFF_, (size_t)BT * DFF_);
        gemm(actbf, dw, retr, nullptr, BT, DD, DFF_);

        k_concat<<<cdiv((size_t)BT * 2 * DD, 256), 256, 0, stream>>>(
            state, DD, retr, DD, zb, BT, DD);
        float* outs[3] = { fob, gob, aob };
        for (int m = 0; m < 3; ++m) {
            gemm(zb, W1bf[m], hidb, b1f32[m], BT, HH, 2 * DD);
            k_silu_bf<<<cdiv((size_t)BT * HH, 256), 256, 0, stream>>>(hidb, hbfb,
                                                                      (size_t)BT * HH);
            gemm(hbfb, W2bf[m], outs[m], b2f32[m], BT, DD, HH);
        }
        k_cfc_combine<<<cdiv((size_t)BT * DD, 256), 256, 0, stream>>>(
            fob, gob, aob, state, state, nullptr, 0, DD, (size_t)BT * DD);
    }

    // ---- final norm + tied-embedding logits ----
    k_rmsnorm<<<BT, 256, 0, stream>>>(state, final_norm, finbf, DD);
    gemm(finbf, embbf, (float*)d_out, nullptr, BT, VV, DD);
}
